// HardAndLayer_4114578670423
// MI455X (gfx1250) — 8-bit matrix operand order repaired and checked by reading, compiled, not run on hardware
//
#include <hip/hip_runtime.h>

typedef __attribute__((ext_vector_type(8))) int v8i;
typedef __attribute__((ext_vector_type(4))) unsigned int v4u;

#define D_DIM 8192
#define L_DIM 8192
#define ROWS_PER_BLOCK 16
#define WAVES_PER_BLOCK 8
#define THREADS (WAVES_PER_BLOCK * 32)
#define K_PER_WAVE (D_DIM / WAVES_PER_BLOCK)
#define STEPS (K_PER_WAVE / 64)

__device__ __forceinline__ unsigned int pack01(v4u v) {
  unsigned int t0 = __builtin_amdgcn_perm(v.y, v.x, 0x0C0C0400u);
  unsigned int t1 = __builtin_amdgcn_perm(v.w, v.z, 0x0C0C0400u);
  return __builtin_amdgcn_perm(t1, t0, 0x05040100u);
}

__global__ __launch_bounds__(THREADS) void hard_and_wmma_kernel(
    const int* __restrict__ x, const int* __restrict__ w,
    float* __restrict__ out) {
  __shared__ __align__(16) unsigned char notx[D_DIM];
  __shared__ int partial[WAVES_PER_BLOCK][16];

  const int tid = threadIdx.x;

  {
    const int base = tid * (D_DIM / THREADS);
#pragma unroll
    for (int j = 0; j < 8; ++j) {
      const v4u v = ((const v4u*)(x + base))[j];
      unsigned int p = pack01(v) ^ 0x01010101u;
      ((unsigned int*)notx)[(base >> 2) + j] = p;
    }
  }
  __syncthreads();

  const int wave = tid >> 5;
  const int lane = tid & 31;
  const int m = lane & 15;
  const int h = lane >> 4;
  const long row = (long)blockIdx.x * ROWS_PER_BLOCK + m;
  const int* wrow = w + row * (long)D_DIM;
  const int kwave = wave * K_PER_WAVE;

  v8i acc = {};

  for (int s = 0; s < STEPS; ++s) {
    const int kk = kwave + s * 64;

    v8i a;
#pragma unroll
    for (int g = 0; g < 4; ++g) {
      const int kb = kk + (g >> 1) * 32 + h * 16 + (g & 1) * 8;
      const v4u w0 = __builtin_nontemporal_load((const v4u*)(wrow + kb));
      const v4u w1 = __builtin_nontemporal_load((const v4u*)(wrow + kb + 4));
      a[2 * g] = (int)pack01(w0);
      a[2 * g + 1] = (int)pack01(w1);
    }

    v8i b;
    const v4u b0 = *(const v4u*)(notx + kk + h * 16);
    const v4u b1 = *(const v4u*)(notx + kk + 32 + h * 16);
    b[0] = (int)b0.x; b[1] = (int)b0.y; b[2] = (int)b0.z; b[3] = (int)b0.w;
    b[4] = (int)b1.x; b[5] = (int)b1.y; b[6] = (int)b1.z; b[7] = (int)b1.w;

    acc = __builtin_amdgcn_wmma_i32_16x16x64_iu8(
         false, a,  false, b, acc,
         false,  false);
  }

  if (lane == 0) {
#pragma unroll
    for (int r = 0; r < 8; ++r) partial[wave][r] = acc[r];
  } else if (lane == 16) {
#pragma unroll
    for (int r = 0; r < 8; ++r) partial[wave][8 + r] = acc[r];
  }
  __syncthreads();

  if (tid < 16) {
    int sum = 0;
#pragma unroll
    for (int wv = 0; wv < WAVES_PER_BLOCK; ++wv) sum += partial[wv][tid];
    out[blockIdx.x * ROWS_PER_BLOCK + tid] = (sum == 0) ? 1.0f : 0.0f;
  }
}

extern "C" void kernel_launch(void* const* d_in, const int* in_sizes, int n_in,
                              void* d_out, int out_size, void* d_ws,
                              size_t ws_size, hipStream_t stream) {
  const int* x = (const int*)d_in[0];
  const int* w = (const int*)d_in[1];
  float* out = (float*)d_out;
  (void)in_sizes; (void)n_in; (void)out_size; (void)d_ws; (void)ws_size;

  hard_and_wmma_kernel<<<L_DIM / ROWS_PER_BLOCK, THREADS, 0, stream>>>(x, w, out);
}
